// FourierKARTLayer_38749194944722
// MI455X (gfx1250) — matrix operand order repaired and checked by reading, not compiled, not run on hardware
//
#include <hip/hip_runtime.h>

typedef __attribute__((ext_vector_type(16))) _Float16 v16h;
typedef __attribute__((ext_vector_type(8)))  float    v8f;

#define B_SZ  256
#define D_IN  512
#define D_OUT 64

// Workspace layout: C matrix (256*4 f32) at offset 0, P (64*32 f16) at offset 4096.

// ---------------------------------------------------------------------------
// Precompute kernel:
//   C[b,q]  = X0[b,:] . Wc_w[q,:] + Wc_b[q]                  (1024 dots of 512)
//   P[d,qk] = A[d,q,k]*cos(Bp[d,q,k])  (qk<16, sin-features)
//           = A[d,q,k]*sin(Bp[d,q,k])  (qk>=16, cos-features)
// ---------------------------------------------------------------------------
__global__ void __launch_bounds__(256) fkart_pre(
    const float* __restrict__ X0,
    const float* __restrict__ Wc_w,
    const float* __restrict__ Wc_b,
    const float* __restrict__ A,
    const float* __restrict__ Bp,
    float* __restrict__ Cmat,
    _Float16* __restrict__ P)
{
    int idx = blockIdx.x * blockDim.x + threadIdx.x;
    if (idx < B_SZ * 4) {
        int b = idx >> 2, q = idx & 3;
        const float* xr = X0 + b * D_IN;
        const float* wr = Wc_w + q * D_IN;
        float acc = 0.f;
        #pragma unroll 8
        for (int d = 0; d < D_IN; ++d) acc += xr[d] * wr[d];
        Cmat[idx] = acc + Wc_b[q];
    } else if (idx < B_SZ * 4 + D_OUT * 32) {
        int p  = idx - B_SZ * 4;
        int d  = p >> 5, kk = p & 31;
        int qk = kk & 15;                       // q*4 + k
        float a  = A[d * 16 + qk];
        float bp = Bp[d * 16 + qk];
        float coef = (kk < 16) ? a * __cosf(bp) : a * __sinf(bp);
        P[d * 32 + kk] = (_Float16)coef;
    }
}

// sin/cos of k*theta for k=1..4 via double/triple-angle recurrence:
// only 2 transcendental ops per theta.
__device__ __forceinline__ void sin_series(float th, float* s, float* c)
{
    float s1 = __sinf(th), c1 = __cosf(th);
    float s2 = 2.f * s1 * c1;
    float c2 = c1 * c1 - s1 * s1;
    s[0] = s1;               c[0] = c1;
    s[1] = s2;               c[1] = c2;
    s[2] = s1 * c2 + c1 * s2; c[2] = c1 * c2 - s1 * s2;
    s[3] = 2.f * s2 * c2;     c[3] = c2 * c2 - s2 * s2;
}

// ---------------------------------------------------------------------------
// Main kernel: V (65536 x 64) = F (65536 x 32, built in registers) @ P^T.
// One wave per 16-row M-tile (fixed i, 16 consecutive j); 4 WMMAs per wave.
// ---------------------------------------------------------------------------
__global__ void __launch_bounds__(256) fkart_main(
    const float* __restrict__ t,
    const float* __restrict__ w,
    const float* __restrict__ Cmat,
    const _Float16* __restrict__ P,
    float* __restrict__ V)
{
    const int lane = threadIdx.x & 31;
    const int wave = threadIdx.x >> 5;
    const int tile = blockIdx.x * 8 + wave;   // 0..4095
    const int m0   = tile << 4;               // first of 16 rows (m = i*256 + j)
    const int i    = m0 >> 8;
    const int r    = lane & 15;
    const int hs   = lane >> 4;               // A half-select: 0 -> K 0-7/16-23, 1 -> K 8-15/24-31
    const int j    = (m0 & 255) + r;
    const int qb   = hs * 2;                  // this lane's q-pair: {0,1} or {2,3}

    const float tv  = t[i];
    const float th0 = Cmat[j * 4 + qb]     + w[qb]     * tv;
    const float th1 = Cmat[j * 4 + qb + 1] + w[qb + 1] * tv;

    float s0[4], c0[4], s1v[4], c1v[4];
    sin_series(th0, s0, c0);
    sin_series(th1, s1v, c1v);

    // A-matrix (16x32 f16) per ISA layout: lanes 0-15 pack K=0..7 then 16..23;
    // lanes 16-31 pack K=8..15 then 24..31. Feature K = q*4+k (sin) / 16+q*4+k (cos).
    v16h a = {};
    #pragma unroll
    for (int k = 0; k < 4; ++k) {
        a[k]      = (_Float16)s0[k];     // sin, q = qb
        a[4 + k]  = (_Float16)s1v[k];    // sin, q = qb+1
        a[8 + k]  = (_Float16)c0[k];     // cos, q = qb
        a[12 + k] = (_Float16)c1v[k];    // cos, q = qb+1
    }

    // Four N-tiles of 16 columns; B (32x16 f16): lane holds column n0+r,
    // K=0..7 then 16..23 (lanes 0-15) or K=8..15 then 24..31 (lanes 16-31) of P row.
    v8f acc[4];
    #pragma unroll
    for (int nt = 0; nt < 4; ++nt) {
        const _Float16* pr = P + (nt * 16 + r) * 32 + hs * 8;
        v16h b;
        #pragma unroll
        for (int s = 0; s < 8; ++s) { b[s] = pr[s]; b[8 + s] = pr[16 + s]; }
        v8f z = {};
        acc[nt] = __builtin_amdgcn_wmma_f32_16x16x32_f16(
            /*neg_a=*/false, a, /*neg_b=*/false, b,
            /*c_mod=*/(short)0, z, /*reuse_a=*/false, /*reuse_b=*/false);
    }

    // D layout: VGPR v, lanes 0-15 -> M=v, lanes 16-31 -> M=v+8; N = r.
    // Inner nt loop => 4 x 64B contiguous stores per row (coalesces to 256B/row).
    float* outbase = V + (size_t)(m0 + hs * 8) * 64 + r;
    #pragma unroll
    for (int v = 0; v < 8; ++v) {
        #pragma unroll
        for (int nt = 0; nt < 4; ++nt)
            __builtin_nontemporal_store(acc[nt][v], outbase + v * 64 + nt * 16);
    }
}

extern "C" void kernel_launch(void* const* d_in, const int* in_sizes, int n_in,
                              void* d_out, int out_size, void* d_ws, size_t ws_size,
                              hipStream_t stream)
{
    const float* X0   = (const float*)d_in[0];
    const float* t    = (const float*)d_in[1];
    const float* Wc_w = (const float*)d_in[2];
    const float* Wc_b = (const float*)d_in[3];
    const float* w    = (const float*)d_in[4];
    const float* A    = (const float*)d_in[5];
    const float* Bp   = (const float*)d_in[6];
    float* V = (float*)d_out;

    float*    Cmat = (float*)d_ws;
    _Float16* P    = (_Float16*)((char*)d_ws + 4096);

    // 1024 C-dots + 2048 P-coeffs = 3072 work items -> 12 blocks of 256.
    hipLaunchKernelGGL(fkart_pre, dim3(12), dim3(256), 0, stream,
                       X0, Wc_w, Wc_b, A, Bp, Cmat, P);

    // 65536 output rows / 16 per wave / 8 waves per block = 512 blocks.
    hipLaunchKernelGGL(fkart_main, dim3(512), dim3(256), 0, stream,
                       t, w, Cmat, P, V);
}
